// SurroundOccPredictionLite_81183471829048
// MI455X (gfx1250) — matrix operand order repaired and checked by reading, not compiled, not run on hardware
//
#include <hip/hip_runtime.h>
#include <hip/hip_bf16.h>

// ---------- problem constants (fixed by setup_inputs) ----------
#define NCAM 6
#define CIN  256
#define EDIM 256
#define HF   29
#define WF   50
#define HW   (HF * WF)      // 1450
#define HB   100
#define WB   100
#define QN   (HB * WB)      // 10000
#define DBIN 4

typedef __attribute__((ext_vector_type(16))) __bf16 v16bf;
typedef __attribute__((ext_vector_type(8)))  float  v8f;

__device__ __forceinline__ __bf16 f2bf(float x) { return (__bf16)x; }

// =====================================================================
// Kernel 1: per-camera channel projection GEMM (bf16 WMMA, f32 accum)
//   feats[n][o][hw] = sum_c proj_w[o][c] * X[n][c][hw]
// stored transposed as featsT[n][hw][o] for coalesced channel sampling.
// One wave computes a 16x32 output slab (two 16x16 tiles, shared A).
// K = 256 fully unrolled -> 16x v_wmma_f32_16x16x32_bf16 per wave.
// B loads are UNCONDITIONAL (clamped address): an out-of-range column
// only pollutes the matching D column, whose store is guarded anyway.
// =====================================================================
__global__ __launch_bounds__(32)
void proj_gemm(const float* __restrict__ X,      // [NCAM][CIN][HW]
               const float* __restrict__ W,      // [EDIM][CIN]
               float* __restrict__ featsT)       // [NCAM][HW][EDIM]
{
    const int lane  = threadIdx.x;          // 0..31 (wave32)
    const int m     = lane & 15;            // row/col within tile
    const int hi    = lane >> 4;            // half-wave select
    const int ntile = blockIdx.x;           // spatial slab of 32 (46)
    const int mtile = blockIdx.y;           // out-channel tile (16)
    const int cam   = blockIdx.z;           // camera (6)

    const float* __restrict__ Wrow = W + (size_t)(mtile * 16 + m) * CIN;
    const float* __restrict__ Xc   = X + (size_t)cam * CIN * HW;

    const int n0 = ntile * 32 + m;          // column of tile 0
    const int n1 = n0 + 16;                 // column of tile 1
    // clamped (always in-bounds) load addresses -> no per-load predication
    const int nc0 = n0 < HW ? n0 : (HW - 1);
    const int nc1 = n1 < HW ? n1 : (HW - 1);
    const float* __restrict__ Xcol0 = Xc + nc0;
    const float* __restrict__ Xcol1 = Xc + nc1;

    v8f acc0 = {};
    v8f acc1 = {};
#pragma unroll
    for (int k0 = 0; k0 < CIN; k0 += 32) {
        // --- A fragment (16x32 bf16): lane<16 holds K = 0..15,
        //     lane>=16 holds K = 16..31 (contiguous, matches B)
        v16bf a;
        const int abase = k0 + (hi ? 16 : 0);
#pragma unroll
        for (int j = 0; j < 8; ++j) {
            a[j]     = f2bf(Wrow[abase + j]);
            a[j + 8] = f2bf(Wrow[abase + 8 + j]);
        }
        // --- B fragments (32x16 bf16): lane holds column n = lane&15,
        //     contiguous K run (0..15 / 16..31) per half-wave
        const int kb = k0 + (hi ? 16 : 0);
        v16bf b0, b1;
#pragma unroll
        for (int j = 0; j < 16; ++j) {
            b0[j] = f2bf(Xcol0[(size_t)(kb + j) * HW]);
            b1[j] = f2bf(Xcol1[(size_t)(kb + j) * HW]);
        }
        acc0 = __builtin_amdgcn_wmma_f32_16x16x32_bf16(
                   false, a, false, b0, (short)0, acc0, false, false);
        acc1 = __builtin_amdgcn_wmma_f32_16x16x32_bf16(
                   false, a, false, b1, (short)0, acc1, false, false);
    }

    // --- D layout: lane holds column N = lane&15, rows M = hi*8 + r.
    // consecutive r = consecutive output channel -> contiguous store.
    const int ochan = mtile * 16 + hi * 8;
    if (n0 < HW) {
        float* dst = featsT + ((size_t)cam * HW + n0) * EDIM + ochan;
#pragma unroll
        for (int r = 0; r < 8; ++r) dst[r] = acc0[r];
    }
    if (n1 < HW) {
        float* dst = featsT + ((size_t)cam * HW + n1) * EDIM + ochan;
#pragma unroll
        for (int r = 0; r < 8; ++r) dst[r] = acc1[r];
    }
}

// =====================================================================
// Kernel 2: BEV geometry + bilinear sampling + masked mean over (n,d)
// One block (256 threads) per BEV query; thread e owns channel e.
// Control flow is uniform across the block (validity is per (n,d)).
// =====================================================================
__global__ __launch_bounds__(256)
void sample_fuse(const float* __restrict__ featsT,   // [NCAM][HW][EDIM]
                 const float* __restrict__ l2i,      // [1][NCAM][4][4]
                 const float* __restrict__ imghw,    // [1][NCAM][2]
                 float* __restrict__ fused)          // [QN][EDIM]
{
    const int q  = blockIdx.x;
    const int e  = threadIdx.x;
    const int yb = q / WB;
    const int xb = q - yb * WB;

    // normalized -> real coords (PC_RANGE = [-50,-50,-5, 50,50,3])
    const float rx = ((xb + 0.5f) / (float)WB) * 100.0f - 50.0f;
    const float ry = ((yb + 0.5f) / (float)HB) * 100.0f - 50.0f;

    float num = 0.0f;
    float den = 0.0f;

    for (int d = 0; d < DBIN; ++d) {
        const float rz = ((d + 0.5f) / (float)DBIN) * 8.0f - 5.0f;
        for (int cam = 0; cam < NCAM; ++cam) {
            const float* L = l2i + cam * 16;
            const float c0 = L[0] * rx + L[1] * ry + L[2]  * rz + L[3];
            const float c1 = L[4] * rx + L[5] * ry + L[6]  * rz + L[7];
            const float c2 = L[8] * rx + L[9] * ry + L[10] * rz + L[11];
            const float dd = fmaxf(c2, 1e-5f);
            const float ih = imghw[cam * 2 + 0];
            const float iw = imghw[cam * 2 + 1];
            const float px = (c0 / dd) / iw;
            const float py = (c1 / dd) / ih;
            const bool valid = (c2 > 1e-5f) &&
                               (px > 0.0f) && (px < 1.0f) &&
                               (py > 0.0f) && (py < 1.0f);
            if (!valid) continue;          // uniform across the block

            const float gx = px * (float)WF - 0.5f;
            const float gy = py * (float)HF - 0.5f;
            const float x0f = floorf(gx), y0f = floorf(gy);
            const float wx = gx - x0f, wy = gy - y0f;
            const int x0 = (int)x0f, y0 = (int)y0f;
            const int x1 = x0 + 1,   y1 = y0 + 1;
            const float w00 = (1.0f - wx) * (1.0f - wy);
            const float w10 = wx * (1.0f - wy);
            const float w01 = (1.0f - wx) * wy;
            const float w11 = wx * wy;

            const float* base = featsT + (size_t)cam * HW * EDIM;
            float s = 0.0f;
            if (x0 >= 0 && y0 >= 0) {
                if (e == 0) __builtin_prefetch(base + (y0 * WF + x0) * EDIM, 0, 0);
                s += base[(size_t)(y0 * WF + x0) * EDIM + e] * w00;
            }
            if (x1 < WF && y0 >= 0)
                s += base[(size_t)(y0 * WF + x1) * EDIM + e] * w10;
            if (x0 >= 0 && y1 < HF)
                s += base[(size_t)(y1 * WF + x0) * EDIM + e] * w01;
            if (x1 < WF && y1 < HF)
                s += base[(size_t)(y1 * WF + x1) * EDIM + e] * w11;

            num += s;
            den += 1.0f;
        }
    }

    fused[(size_t)q * EDIM + e] = num / fmaxf(den, 1.0f);
}

// =====================================================================
// Kernel 3: output GEMM: out_pre[q][e] = bq[e] + sum_k fused[q][k]*Wq[e][k]
// bf16 WMMA; one wave computes a 16x32 slab (two 16x16 tiles, shared A).
// Q = 625*16 exactly, E = 8*32 exactly -> no guards needed.
// =====================================================================
__global__ __launch_bounds__(32)
void out_gemm(const float* __restrict__ fused,   // [QN][EDIM]
              const float* __restrict__ Wq,      // [EDIM][EDIM]
              const float* __restrict__ bq,      // [EDIM]
              float* __restrict__ outpre)        // [QN][EDIM]
{
    const int lane  = threadIdx.x;
    const int m     = lane & 15;
    const int hi    = lane >> 4;
    const int qtile = blockIdx.x;   // 625
    const int etile = blockIdx.y;   // 8 (slabs of 32 output columns)

    const float* __restrict__ Arow  = fused + (size_t)(qtile * 16 + m) * EDIM;
    // B[k][n] = Wq[n][k] -> contiguous k within Wq row n
    const float* __restrict__ Brow0 = Wq + (size_t)(etile * 32 + m) * EDIM;
    const float* __restrict__ Brow1 = Brow0 + (size_t)16 * EDIM;

    v8f acc0 = {};
    v8f acc1 = {};
#pragma unroll
    for (int k0 = 0; k0 < EDIM; k0 += 32) {
        v16bf a;
        const int abase = k0 + (hi ? 16 : 0);
#pragma unroll
        for (int j = 0; j < 8; ++j) {
            a[j]     = f2bf(Arow[abase + j]);
            a[j + 8] = f2bf(Arow[abase + 8 + j]);
        }
        v16bf b0, b1;
        const int kb = k0 + (hi ? 16 : 0);
#pragma unroll
        for (int j = 0; j < 16; ++j) {
            b0[j] = f2bf(Brow0[kb + j]);
            b1[j] = f2bf(Brow1[kb + j]);
        }
        acc0 = __builtin_amdgcn_wmma_f32_16x16x32_bf16(
                   false, a, false, b0, (short)0, acc0, false, false);
        acc1 = __builtin_amdgcn_wmma_f32_16x16x32_bf16(
                   false, a, false, b1, (short)0, acc1, false, false);
    }

    // D: lane column e = etile*32 + (lane&15) [+16]; rows q = qtile*16+hi*8+r
    const int   e0 = etile * 32 + m;
    const int   e1 = e0 + 16;
    const float bias0 = bq[e0];
    const float bias1 = bq[e1];
    float* dst = outpre + (size_t)(qtile * 16 + hi * 8) * EDIM;
#pragma unroll
    for (int r = 0; r < 8; ++r) {
        dst[(size_t)r * EDIM + e0] = acc0[r] + bias0;
        dst[(size_t)r * EDIM + e1] = acc1[r] + bias1;
    }
}

// =====================================================================
// Kernel 4: LayerNorm over E per row (LDS tree reduction)
// =====================================================================
__global__ __launch_bounds__(256)
void layernorm(const float* __restrict__ outpre,  // [QN][EDIM]
               const float* __restrict__ gamma,
               const float* __restrict__ beta,
               float* __restrict__ out)            // [QN][EDIM]
{
    __shared__ float red[EDIM];
    const int q = blockIdx.x;
    const int e = threadIdx.x;

    const float v = outpre[(size_t)q * EDIM + e];
    red[e] = v;
    __syncthreads();
#pragma unroll
    for (int s = EDIM / 2; s > 0; s >>= 1) {
        if (e < s) red[e] += red[e + s];
        __syncthreads();
    }
    const float mu = red[0] * (1.0f / EDIM);
    __syncthreads();

    const float dv = v - mu;
    red[e] = dv * dv;
    __syncthreads();
#pragma unroll
    for (int s = EDIM / 2; s > 0; s >>= 1) {
        if (e < s) red[e] += red[e + s];
        __syncthreads();
    }
    const float var = red[0] * (1.0f / EDIM);

    out[(size_t)q * EDIM + e] =
        gamma[e] * dv * __frsqrt_rn(var + 1e-5f) + beta[e];
}

// =====================================================================
// Host-side launcher (graph-capture safe: only kernel launches on stream)
// =====================================================================
extern "C" void kernel_launch(void* const* d_in, const int* in_sizes, int n_in,
                              void* d_out, int out_size, void* d_ws, size_t ws_size,
                              hipStream_t stream) {
    const float* camf  = (const float*)d_in[0];   // (1,6,256,29,50)
    const float* projw = (const float*)d_in[1];   // (256,256)
    const float* l2i   = (const float*)d_in[2];   // (1,6,4,4)
    const float* imghw = (const float*)d_in[3];   // (1,6,2)
    const float* Wq    = (const float*)d_in[4];   // (256,256)
    const float* bq    = (const float*)d_in[5];   // (256,)
    const float* gamma = (const float*)d_in[6];   // (256,)
    const float* beta  = (const float*)d_in[7];   // (256,)
    float*       out   = (float*)d_out;           // (1,10000,256) f32

    float* featsT = (float*)d_ws;                              // 6*1450*256
    float* fused  = featsT + (size_t)NCAM * HW * EDIM;         // 10000*256
    float* outpre = fused  + (size_t)QN * EDIM;                // 10000*256

    // 1) channel projection GEMM (WMMA bf16), 16x32 slab per wave
    dim3 g1((HW + 31) / 32, EDIM / 16, NCAM);
    proj_gemm<<<g1, 32, 0, stream>>>(camf, projw, featsT);

    // 2) geometry + bilinear sampling + masked mean
    sample_fuse<<<QN, 256, 0, stream>>>(featsT, l2i, imghw, fused);

    // 3) output GEMM + bias (WMMA bf16), 16x32 slab per wave
    dim3 g3(QN / 16, EDIM / 32);
    out_gemm<<<g3, 32, 0, stream>>>(fused, Wq, bq, outpre);

    // 4) LayerNorm
    layernorm<<<QN, 256, 0, stream>>>(outpre, gamma, beta, out);
}
